// VirtualNode_ConGNN_PriorAware_67654324846759
// MI455X (gfx1250) — matrix operand order repaired and checked by reading, not compiled, not run on hardware
//
#include <hip/hip_runtime.h>

// ---------------------------------------------------------------------------
// MI455X (gfx1250) implementation of the VirtualNode ConGNN reference.
// Dense layers: bf16 WMMA GEMM (v_wmma_f32_16x16x32_bf16, f32 accumulate)
// with pre-converted bf16 operands (no cvt / no predication in the K-loop).
// Graph gather/scatter/softmax: bandwidth/atomic kernels.
// ---------------------------------------------------------------------------

typedef __attribute__((ext_vector_type(16))) __bf16 v16bf;
typedef __attribute__((ext_vector_type(8)))  __bf16 v8bf;
typedef __attribute__((ext_vector_type(4)))  __bf16 v4bf;
typedef __attribute__((ext_vector_type(8)))  float  v8f;

#define DEVINL __device__ __forceinline__

DEVINL float warp_sum32(float s) {
#pragma unroll
  for (int o = 16; o > 0; o >>= 1) s += __shfl_xor(s, o, 32);
  return s;
}

DEVINL void atomicMaxF(float* addr, float val) {
  int old = __float_as_int(*addr);
  while (__int_as_float(old) < val) {
    int prev = atomicCAS((int*)addr, old, __float_as_int(val));
    if (prev == old) break;
    old = prev;
  }
}

// ------------------------------- fill / cvt --------------------------------
__global__ void k_fill(float* __restrict__ p, float v, long n) {
  long t = (long)blockIdx.x * blockDim.x + threadIdx.x;
  if (t < n) p[t] = v;
}

// f32 -> bf16, 4 elements per thread (n must be a multiple of 4).
__global__ void k_cvt4(const float* __restrict__ src, __bf16* __restrict__ dst, long n) {
  long i = ((long)blockIdx.x * blockDim.x + threadIdx.x) * 4;
  if (i >= n) return;
  float4 x = *(const float4*)(src + i);
  v4bf y;
  y[0] = (__bf16)x.x; y[1] = (__bf16)x.y; y[2] = (__bf16)x.z; y[3] = (__bf16)x.w;
  *(v4bf*)(dst + i) = y;
}

// strided f32 -> dense bf16 repack (for router_w1 [256,257] -> [256,256])
__global__ void k_cvt_strided(const float* __restrict__ src, int src_ld,
                              __bf16* __restrict__ dst, int rows, int cols) {
  long t = (long)blockIdx.x * blockDim.x + threadIdx.x;
  if (t >= (long)rows * cols) return;
  int r = (int)(t / cols), c = (int)(t - (long)r * cols);
  dst[t] = (__bf16)src[(size_t)r * src_ld + c];
}

// ------------------------- WMMA fragment loads -----------------------------
// A fragment (16x32 bf16, ISA 7.12.2): lane L holds row L%16.
// lanes 0-15: K = k0+0..7 and k0+16..23 ; lanes16-31: K = k0+8..15 and +24..31
DEVINL v16bf load_frag_a(const __bf16* __restrict__ prow, int kb) {
  v8bf lo = *(const v8bf*)(prow + kb);
  v8bf hi = *(const v8bf*)(prow + kb + 16);
  return __builtin_shufflevector(lo, hi, 0, 1, 2, 3, 4, 5, 6, 7,
                                 8, 9, 10, 11, 12, 13, 14, 15);
}

// B fragment (32x16 bf16): lane L holds column L%16 (== weight row).
// lanes 0-15: K = k0+0..7 and k0+16..23 ; lanes16-31: K = k0+8..15 and +24..31
DEVINL v16bf load_frag_b(const __bf16* __restrict__ prow, int kb) {
  v8bf lo = *(const v8bf*)(prow + kb);
  v8bf hi = *(const v8bf*)(prow + kb + 16);
  return __builtin_shufflevector(lo, hi, 0, 1, 2, 3, 4, 5, 6, 7,
                                 8, 9, 10, 11, 12, 13, 14, 15);
}

// C/D layout (16x16 f32): vgpr r, lane L -> M = r + 8*(L/16), N = L%16.
template <int ACT>
DEVINL void epilogue_tile(v8f acc, int m0, int n, int lane,
                          const float* __restrict__ bias,
                          const float* __restrict__ ein, const float* __restrict__ ew, int ews,
                          const float* __restrict__ resid,
                          float* __restrict__ C, __bf16* __restrict__ Cbf,
                          int ldc, int M, int N) {
  if (n >= N) return;
  const float badd = bias ? bias[n] : 0.0f;
  const float wext = ein ? ew[(size_t)n * ews] : 0.0f;
  const int mb = m0 + ((lane >> 4) << 3);
#pragma unroll
  for (int r = 0; r < 8; ++r) {
    int m = mb + r;
    if (m < M) {
      float v = acc[r] + badd;
      if (ein)   v += ein[m] * wext;
      if (resid) v += resid[(size_t)m * ldc + n];
      if (ACT == 1) v = fmaxf(v, 0.0f);
      if (C)   C[(size_t)m * ldc + n] = v;
      if (Cbf) Cbf[(size_t)m * ldc + n] = (__bf16)v;
    }
  }
}

// C[M,N] = act(A[M,K]*W[N,K]^T + bias + ein*ew + resid). One wave = 16x64.
// Requirements (guaranteed by host): K%32==0, lda%8==0, ldw%8==0.
// OOB rows are clamped (their products land only in store-masked outputs).
template <int ACT>
__global__ void k_gemm(const __bf16* __restrict__ A, int lda,
                       const __bf16* __restrict__ W, int ldw,
                       const float* __restrict__ bias,
                       const float* __restrict__ ein, const float* __restrict__ ew, int ews,
                       const float* __restrict__ resid,
                       float* __restrict__ C, __bf16* __restrict__ Cbf, int ldc,
                       int M, int N, int K, int tilesM, int tilesNG) {
  const int wid = blockIdx.x * (blockDim.x >> 5) + (threadIdx.x >> 5);
  if (wid >= tilesM * tilesNG) return;
  const int lane = threadIdx.x & 31;
  const int tm = wid / tilesNG;
  const int tg = wid - tm * tilesNG;
  const int m0 = tm << 4;
  const int n0 = tg << 6;
  const int l15 = lane & 15;
  int arow = m0 + l15;            if (arow >= M) arow = M - 1;
  int br0 = n0 + l15;             if (br0 >= N) br0 = N - 1;
  int br1 = n0 + 16 + l15;        if (br1 >= N) br1 = N - 1;
  int br2 = n0 + 32 + l15;        if (br2 >= N) br2 = N - 1;
  int br3 = n0 + 48 + l15;        if (br3 >= N) br3 = N - 1;
  const __bf16* pa  = A + (size_t)arow * lda;
  const __bf16* pb0 = W + (size_t)br0 * ldw;
  const __bf16* pb1 = W + (size_t)br1 * ldw;
  const __bf16* pb2 = W + (size_t)br2 * ldw;
  const __bf16* pb3 = W + (size_t)br3 * ldw;
  const int ka = (lane >> 4) << 3;   // +0 / +8
  const int kb = (lane >> 4) << 3;   // +0 / +8
  v8f acc0 = {}; v8f acc1 = {}; v8f acc2 = {}; v8f acc3 = {};
  for (int k0 = 0; k0 < K; k0 += 32) {
    v16bf a  = load_frag_a(pa, k0 + ka);
    v16bf b0 = load_frag_b(pb0, k0 + kb);
    v16bf b1 = load_frag_b(pb1, k0 + kb);
    v16bf b2 = load_frag_b(pb2, k0 + kb);
    v16bf b3 = load_frag_b(pb3, k0 + kb);
    acc0 = __builtin_amdgcn_wmma_f32_16x16x32_bf16(false, a, false, b0, (short)0, acc0, false, false);
    acc1 = __builtin_amdgcn_wmma_f32_16x16x32_bf16(false, a, false, b1, (short)0, acc1, false, false);
    acc2 = __builtin_amdgcn_wmma_f32_16x16x32_bf16(false, a, false, b2, (short)0, acc2, false, false);
    acc3 = __builtin_amdgcn_wmma_f32_16x16x32_bf16(false, a, false, b3, (short)0, acc3, false, false);
  }
  const int nb = n0 + l15;
  epilogue_tile<ACT>(acc0, m0, nb,      lane, bias, ein, ew, ews, resid, C, Cbf, ldc, M, N);
  epilogue_tile<ACT>(acc1, m0, nb + 16, lane, bias, ein, ew, ews, resid, C, Cbf, ldc, M, N);
  epilogue_tile<ACT>(acc2, m0, nb + 32, lane, bias, ein, ew, ews, resid, C, Cbf, ldc, M, N);
  epilogue_tile<ACT>(acc3, m0, nb + 48, lane, bias, ein, ew, ews, resid, C, Cbf, ldc, M, N);
}

// ----------------------------- GATv2 kernels -------------------------------
DEVINL void edge_sd(const int* __restrict__ ei, int E, int e, int& s, int& d) {
  if (e < E) { s = ei[e]; d = ei[E + e]; } else { s = d = e - E; }
}

__global__ void k_gat_scores(const float* __restrict__ xl, const float* __restrict__ xr,
                             const int* __restrict__ ei, int E, int N,
                             const float* __restrict__ att,
                             float* __restrict__ esc, float* __restrict__ nmax) {
  int t = blockIdx.x * blockDim.x + threadIdx.x;
  int total = (E + N) * 4;
  if (t >= total) return;
  int e = t >> 2, h = t & 3, s, d;
  edge_sd(ei, E, e, s, d);
  const float* pl = xl + (size_t)s * 128 + h * 32;
  const float* pr = xr + (size_t)d * 128 + h * 32;
  const float* pa = att + h * 32;
  float acc = 0.0f;
#pragma unroll 8
  for (int c = 0; c < 32; ++c) {
    float z = pl[c] + pr[c];
    z = z > 0.0f ? z : 0.2f * z;
    acc += z * pa[c];
  }
  esc[t] = acc;
  atomicMaxF(&nmax[d * 4 + h], acc);
}

__global__ void k_gat_exp(float* __restrict__ esc, const float* __restrict__ nmax,
                          float* __restrict__ nsum, const int* __restrict__ ei, int E, int N) {
  int t = blockIdx.x * blockDim.x + threadIdx.x;
  int total = (E + N) * 4;
  if (t >= total) return;
  int e = t >> 2, h = t & 3, s, d;
  edge_sd(ei, E, e, s, d);
  float ex = __expf(esc[t] - nmax[d * 4 + h]);
  esc[t] = ex;
  atomicAdd(&nsum[d * 4 + h], ex);
}

__global__ void k_gat_accum(const float* __restrict__ xl, const float* __restrict__ esc,
                            const float* __restrict__ nsum, const int* __restrict__ ei,
                            int E, int N, float* __restrict__ gout) {
  int t = blockIdx.x * blockDim.x + threadIdx.x;
  int total = (E + N) * 4;
  if (t >= total) return;
  int e = t >> 2, h = t & 3, s, d;
  edge_sd(ei, E, e, s, d);
  float a = esc[t] / (nsum[d * 4 + h] + 1e-16f);
  const float* pl = xl + (size_t)s * 128 + h * 32;
  float* po = gout + (size_t)d * 128 + h * 32;
#pragma unroll 8
  for (int c = 0; c < 32; ++c) atomicAdd(&po[c], pl[c] * a);
}

// h = LayerNorm(h + elu(gout + gb)); one wave per node; dual f32+bf16 store.
__global__ void k_gat_post(float* __restrict__ h, __bf16* __restrict__ hbf,
                           const float* __restrict__ gout,
                           const float* __restrict__ gb, const float* __restrict__ gamma,
                           const float* __restrict__ beta, int N) {
  int node = blockIdx.x * (blockDim.x >> 5) + (threadIdx.x >> 5);
  int lane = threadIdx.x & 31;
  if (node >= N) return;
  float y[4];
  float s = 0.0f;
#pragma unroll
  for (int i = 0; i < 4; ++i) {
    int c = i * 32 + lane;
    float g = gout[(size_t)node * 128 + c] + gb[c];
    g = g > 0.0f ? g : (__expf(g) - 1.0f);
    y[i] = h[(size_t)node * 128 + c] + g;
    s += y[i];
  }
  float mean = warp_sum32(s) * (1.0f / 128.0f);
  float v = 0.0f;
#pragma unroll
  for (int i = 0; i < 4; ++i) { float dd = y[i] - mean; v += dd * dd; }
  float var = warp_sum32(v) * (1.0f / 128.0f);
  float inv = rsqrtf(var + 1e-5f);
#pragma unroll
  for (int i = 0; i < 4; ++i) {
    int c = i * 32 + lane;
    float o = (y[i] - mean) * inv * gamma[c] + beta[c];
    h[(size_t)node * 128 + c] = o;
    hbf[(size_t)node * 128 + c] = (__bf16)o;
  }
}

// ------------------------- pooling + classifiers ---------------------------
__global__ void k_pool_accum(const float* __restrict__ h, const int* __restrict__ batch,
                             float* __restrict__ pool, float* __restrict__ cnt, long N, int D) {
  long t = (long)blockIdx.x * blockDim.x + threadIdx.x;
  if (t >= N * D) return;
  long n = t / D;
  int d = (int)(t - n * D);
  int g = batch[n];
  atomicAdd(&pool[(size_t)g * D + d], h[t]);
  if (d == 0) atomicAdd(&cnt[g], 1.0f);
}

__global__ void k_pool_clf(const float* __restrict__ pool, const float* __restrict__ cnt,
                           const float* __restrict__ w, const float* __restrict__ b,
                           float* __restrict__ out, int G, int D) {
  int t = blockIdx.x * blockDim.x + threadIdx.x;
  if (t >= G * 3) return;
  int g = t / 3, o = t - g * 3;
  float c = fmaxf(cnt[g], 1.0f);
  float acc = 0.0f;
  for (int d = 0; d < D; ++d) acc += pool[(size_t)g * D + d] * w[(size_t)o * D + d];
  out[t] = acc / c + b[o];
}

// ------------------------------ IPR kernels --------------------------------
// top_k(4,3) keeps everything except the minimum (ties: drop highest index).
__global__ void k_topk(const float* __restrict__ logits, float* __restrict__ wm, long N) {
  long n = (long)blockIdx.x * blockDim.x + threadIdx.x;
  if (n >= N) return;
  const float* l = logits + n * 4;
  int ex = 0; float mv = l[0];
#pragma unroll
  for (int v = 1; v < 4; ++v) if (l[v] <= mv) { mv = l[v]; ex = v; }
  float* w = wm + n * 4;
#pragma unroll
  for (int v = 0; v < 4; ++v) w[v] = (v == ex) ? 0.0f : 1.0f;
}

__global__ void k_vn_scatter(const float* __restrict__ r2v, const float* __restrict__ wm,
                             const int* __restrict__ rb, float* __restrict__ msgs, long N) {
  long t = (long)blockIdx.x * blockDim.x + threadIdx.x;
  if (t >= N * 256) return;
  long n = t >> 8;
  int d = (int)(t & 255);
  int g = rb[n];
  float m = r2v[t];
  const float* w = wm + n * 4;
#pragma unroll
  for (int v = 0; v < 4; ++v)
    if (w[v] > 0.5f) atomicAdd(&msgs[((size_t)g * 4 + v) * 256 + d], m);
}

__global__ void k_vn_build(float* __restrict__ vn, __bf16* __restrict__ vnbf,
                           const float* __restrict__ vn_init,
                           const float* __restrict__ msgs, int G) {
  long t = (long)blockIdx.x * blockDim.x + threadIdx.x;
  if (t >= (long)G * 1024) return;
  float v = vn_init[t & 1023] + msgs[t];
  vn[t] = v;
  vnbf[t] = (__bf16)v;
}

// Self-attention over 4 VNs: thread per (g, head, query row).
__global__ void k_mha(const float* __restrict__ qkv, float* __restrict__ o,
                      __bf16* __restrict__ obf, int G) {
  int t = blockIdx.x * blockDim.x + threadIdx.x;
  if (t >= G * 16) return;
  int g = t >> 4, h = (t >> 2) & 3, q = t & 3;
  const float* Q = qkv + ((size_t)(g * 4 + q)) * 768 + h * 64;
  float sc[4];
#pragma unroll
  for (int kk = 0; kk < 4; ++kk) {
    const float* Kp = qkv + ((size_t)(g * 4 + kk)) * 768 + 256 + h * 64;
    float a = 0.0f;
    for (int d = 0; d < 64; ++d) a += Q[d] * Kp[d];
    sc[kk] = a * 0.125f;  // 1/sqrt(64)
  }
  float m = fmaxf(fmaxf(sc[0], sc[1]), fmaxf(sc[2], sc[3]));
  float ssum = 0.0f;
#pragma unroll
  for (int kk = 0; kk < 4; ++kk) { sc[kk] = __expf(sc[kk] - m); ssum += sc[kk]; }
  float invs = 1.0f / ssum;
  size_t ob = ((size_t)(g * 4 + q)) * 256 + h * 64;
  for (int d = 0; d < 64; ++d) {
    float acc = 0.0f;
#pragma unroll
    for (int kk = 0; kk < 4; ++kk)
      acc += sc[kk] * qkv[((size_t)(g * 4 + kk)) * 768 + 512 + h * 64 + d];
    float v = acc * invs;
    o[ob + d] = v;
    obf[ob + d] = (__bf16)v;
  }
}

__global__ void k_realmsg(const float* __restrict__ vn2, const float* __restrict__ wm,
                          const int* __restrict__ rb, float* __restrict__ rm,
                          __bf16* __restrict__ rmbf, long N) {
  long t = (long)blockIdx.x * blockDim.x + threadIdx.x;
  if (t >= N * 256) return;
  long n = t >> 8;
  int d = (int)(t & 255);
  int g = rb[n];
  const float* w = wm + n * 4;
  const float* base = vn2 + (size_t)g * 1024 + d;
  float s = 0.0f;
#pragma unroll
  for (int v = 0; v < 4; ++v) if (w[v] > 0.5f) s += base[(size_t)v * 256];
  rm[t] = s;
  rmbf[t] = (__bf16)s;
}

__global__ void k_gru(const float* __restrict__ gi, const float* __restrict__ gh,
                      const float* __restrict__ rn, float* __restrict__ upd, long N) {
  long t = (long)blockIdx.x * blockDim.x + threadIdx.x;
  if (t >= N * 256) return;
  long n = t >> 8;
  int d = (int)(t & 255);
  const float* pi = gi + n * 768;
  const float* ph = gh + n * 768;
  float r  = 1.0f / (1.0f + __expf(-(pi[d]       + ph[d])));
  float z  = 1.0f / (1.0f + __expf(-(pi[256 + d] + ph[256 + d])));
  float nn = tanhf(pi[512 + d] + r * ph[512 + d]);
  upd[t] = (1.0f - z) * nn + z * rn[t];
}

// ------------------------------- host side ---------------------------------
static inline int cdiv(long a, long b) { return (int)((a + b - 1) / b); }

static void gemm(hipStream_t st, const __bf16* A, int lda, const __bf16* W, int ldw,
                 const float* bias, const float* ein, const float* ew, int ews,
                 const float* resid, float* C, __bf16* Cbf, int ldc,
                 int M, int N, int K, bool relu) {
  int tilesM = (M + 15) >> 4;
  int tilesNG = (N + 63) >> 6;
  long waves = (long)tilesM * tilesNG;
  int blocks = (int)((waves + 3) >> 2);  // 4 waves / 128-thread block
  if (relu)
    k_gemm<1><<<blocks, 128, 0, st>>>(A, lda, W, ldw, bias, ein, ew, ews, resid, C, Cbf, ldc, M, N, K, tilesM, tilesNG);
  else
    k_gemm<0><<<blocks, 128, 0, st>>>(A, lda, W, ldw, bias, ein, ew, ews, resid, C, Cbf, ldc, M, N, K, tilesM, tilesNG);
}

extern "C" void kernel_launch(void* const* d_in, const int* in_sizes, int n_in,
                              void* d_out, int out_size, void* d_ws, size_t ws_size,
                              hipStream_t stream) {
  const int FD = 2048, G = 256;
  const int NF = in_sizes[0] / FD;
  const int EF = in_sizes[1] / 2;
  const int NC = in_sizes[4] / FD;
  const int EC = in_sizes[5] / 2;
  const long NA = (long)NF + NC;
  const int NMX = NF > NC ? NF : NC;
  const int EMX = EF > EC ? EF : EC;

  const float* face_x  = (const float*)d_in[0];
  const int*   face_ei = (const int*)  d_in[1];
  const int*   face_b  = (const int*)  d_in[2];
  const float* face_pr = (const float*)d_in[3];
  const float* ctx_x   = (const float*)d_in[4];
  const int*   ctx_ei  = (const int*)  d_in[5];
  const int*   ctx_b   = (const int*)  d_in[6];
  const float* ctx_pr  = (const float*)d_in[7];
  // d_in[8] = num_graphs (256, fixed by setup)

  int p = 9;
  struct GatP {
    const float *in_w, *in_b, *wl, *bl, *wr, *br, *att, *gb, *lng, *lnb;
  };
  auto take = [&](void) { return (const float*)d_in[p++]; };
  GatP fp, cp;
  fp.in_w = take(); fp.in_b = take(); fp.wl = take(); fp.bl = take(); fp.wr = take();
  fp.br = take(); fp.att = take(); fp.gb = take(); fp.lng = take(); fp.lnb = take();
  cp.in_w = take(); cp.in_b = take(); cp.wl = take(); cp.bl = take(); cp.wr = take();
  cp.br = take(); cp.att = take(); cp.gb = take(); cp.lng = take(); cp.lnb = take();
  const float* proj_w    = take(); const float* proj_b    = take();
  const float* vn_init   = take();
  const float* router_w1 = take(); const float* router_b1 = take();
  const float* router_w2 = take(); const float* router_b2 = take();
  const float* msg_w     = take(); const float* msg_b     = take();
  const float* mha_in_w  = take(); const float* mha_in_b  = take();
  const float* mha_out_w = take(); const float* mha_out_b = take();
  const float* gru_w_ih  = take(); const float* gru_w_hh  = take();
  const float* gru_b_ih  = take(); const float* gru_b_hh  = take();
  const float* clf_f_w   = take(); const float* clf_f_b   = take();
  const float* clf_c_w   = take(); const float* clf_c_b   = take();
  const float* clf_w_w   = take(); const float* clf_w_b   = take();

  // ---- workspace (bump allocator, 256B aligned) ----
  char* wsp = (char*)d_ws;
  size_t off = 0;
  auto alloc = [&](size_t bytes) -> void* {
    void* r = wsp + off;
    off = (off + bytes + 255) & ~(size_t)255;
    return r;
  };
  float*  h      = (float*) alloc((size_t)NA * 128 * 4);  // hf rows [0,NF), hc rows [NF,NA)
  __bf16* hbf    = (__bf16*)alloc((size_t)NA * 128 * 2);
  float*  rn     = (float*) alloc((size_t)NA * 256 * 4);
  __bf16* rnbf   = (__bf16*)alloc((size_t)NA * 256 * 2);
  int*    rb     = (int*)   alloc((size_t)NA * 4);
  float*  pr     = (float*) alloc((size_t)NA * 4);
  float*  pool   = (float*) alloc((size_t)G * 256 * 4);
  float*  cnt    = (float*) alloc((size_t)G * 4);
  float*  wm     = (float*) alloc((size_t)NA * 4 * 4);
  float*  logits = (float*) alloc((size_t)NA * 4 * 4);
  float*  msgs   = (float*) alloc((size_t)G * 1024 * 4);
  float*  vn     = (float*) alloc((size_t)G * 1024 * 4);
  __bf16* vnbf   = (__bf16*)alloc((size_t)G * 1024 * 2);
  float*  qkv    = (float*) alloc((size_t)G * 4 * 768 * 4);
  float*  attno  = (float*) alloc((size_t)G * 1024 * 4);
  __bf16* attnbf = (__bf16*)alloc((size_t)G * 1024 * 2);
  float*  vn2    = (float*) alloc((size_t)G * 1024 * 4);
  float*  r2v    = (float*) alloc((size_t)NA * 256 * 4);
  float*  rm     = (float*) alloc((size_t)NA * 256 * 4);  // real_msg; later `updated`
  __bf16* rmbf   = (__bf16*)alloc((size_t)NA * 256 * 2);
  // bf16 weight copies
  __bf16* w_fin  = (__bf16*)alloc((size_t)128 * FD * 2);
  __bf16* w_cin  = (__bf16*)alloc((size_t)128 * FD * 2);
  __bf16* w_fwl  = (__bf16*)alloc((size_t)2 * 128 * 128 * 2);
  __bf16* w_fwr  = (__bf16*)alloc((size_t)2 * 128 * 128 * 2);
  __bf16* w_cwl  = (__bf16*)alloc((size_t)2 * 128 * 128 * 2);
  __bf16* w_cwr  = (__bf16*)alloc((size_t)2 * 128 * 128 * 2);
  __bf16* w_proj = (__bf16*)alloc((size_t)256 * 128 * 2);
  __bf16* w_r1   = (__bf16*)alloc((size_t)256 * 256 * 2);  // repacked, stride 256
  __bf16* w_r2   = (__bf16*)alloc((size_t)4 * 256 * 2);
  __bf16* w_msg  = (__bf16*)alloc((size_t)256 * 256 * 2);
  __bf16* w_min  = (__bf16*)alloc((size_t)768 * 256 * 2);
  __bf16* w_mout = (__bf16*)alloc((size_t)256 * 256 * 2);
  __bf16* w_gih  = (__bf16*)alloc((size_t)768 * 256 * 2);
  __bf16* w_ghh  = (__bf16*)alloc((size_t)768 * 256 * 2);
  float*  gi     = (float*) alloc((size_t)NA * 768 * 4);
  float*  gh     = (float*) alloc((size_t)NA * 768 * 4);
  // Per-branch GAT scratch + bf16 input copy alias the (later-used) gi region:
  __bf16* xbf  = (__bf16*)gi;                               // NMX*FD bf16
  float*  xl   = (float*)((char*)gi + (size_t)NMX * FD * 2);
  float*  xr   = xl + (size_t)NMX * 128;
  float*  esc  = xr + (size_t)NMX * 128;
  float*  nmax = esc + (size_t)(EMX + NMX) * 4;
  float*  nsum = nmax + (size_t)NMX * 4;
  float*  gout = nsum + (size_t)NMX * 4;
  __bf16* h1bf = (__bf16*)gi;  // router hidden (bf16 only; before gi is written)
  float*  out  = (float*)d_out;

  // ---- one-time weight conversions ----
  auto cvt = [&](const float* src, __bf16* dst, long n) {
    k_cvt4<<<cdiv(n / 4, 256), 256, 0, stream>>>(src, dst, n);
  };
  cvt(fp.in_w, w_fin, (long)128 * FD);
  cvt(cp.in_w, w_cin, (long)128 * FD);
  cvt(fp.wl, w_fwl, 2 * 128 * 128); cvt(fp.wr, w_fwr, 2 * 128 * 128);
  cvt(cp.wl, w_cwl, 2 * 128 * 128); cvt(cp.wr, w_cwr, 2 * 128 * 128);
  cvt(proj_w, w_proj, 256 * 128);
  k_cvt_strided<<<cdiv(256 * 256, 256), 256, 0, stream>>>(router_w1, 257, w_r1, 256, 256);
  cvt(router_w2, w_r2, 4 * 256);
  cvt(msg_w, w_msg, 256 * 256);
  cvt(mha_in_w, w_min, 768 * 256);
  cvt(mha_out_w, w_mout, 256 * 256);
  cvt(gru_w_ih, w_gih, 768 * 256);
  cvt(gru_w_hh, w_ghh, 768 * 256);

  // ---- GAT branch runner ----
  auto run_branch = [&](const float* X, const int* ei, int E, int Nn, const int* batch,
                        const GatP& q, const __bf16* inw_bf, const __bf16* wl_bf,
                        const __bf16* wr_bf, const float* clf_w, const float* clf_b,
                        float* hout, __bf16* houtbf, float* out3) {
    cvt(X, xbf, (long)Nn * FD);
    gemm(stream, xbf, FD, inw_bf, FD, q.in_b, nullptr, nullptr, 0, nullptr,
         hout, houtbf, 128, Nn, 128, FD, true);
    for (int l = 0; l < 2; ++l) {
      gemm(stream, houtbf, 128, wl_bf + (size_t)l * 128 * 128, 128, q.bl + l * 128,
           nullptr, nullptr, 0, nullptr, xl, nullptr, 128, Nn, 128, 128, false);
      gemm(stream, houtbf, 128, wr_bf + (size_t)l * 128 * 128, 128, q.br + l * 128,
           nullptr, nullptr, 0, nullptr, xr, nullptr, 128, Nn, 128, 128, false);
      long tot = (long)(E + Nn) * 4;
      k_fill<<<cdiv((long)Nn * 4, 256), 256, 0, stream>>>(nmax, -3.0e38f, (long)Nn * 4);
      k_fill<<<cdiv((long)Nn * 4, 256), 256, 0, stream>>>(nsum, 0.0f, (long)Nn * 4);
      k_fill<<<cdiv((long)Nn * 128, 256), 256, 0, stream>>>(gout, 0.0f, (long)Nn * 128);
      k_gat_scores<<<cdiv(tot, 256), 256, 0, stream>>>(xl, xr, ei, E, Nn, q.att + l * 128, esc, nmax);
      k_gat_exp   <<<cdiv(tot, 256), 256, 0, stream>>>(esc, nmax, nsum, ei, E, Nn);
      k_gat_accum <<<cdiv(tot, 256), 256, 0, stream>>>(xl, esc, nsum, ei, E, Nn, gout);
      k_gat_post  <<<cdiv(Nn, 8), 256, 0, stream>>>(hout, houtbf, gout, q.gb + l * 128,
                                                    q.lng + l * 128, q.lnb + l * 128, Nn);
    }
    k_fill<<<cdiv((long)G * 128, 256), 256, 0, stream>>>(pool, 0.0f, (long)G * 128);
    k_fill<<<cdiv((long)G, 256), 256, 0, stream>>>(cnt, 0.0f, (long)G);
    k_pool_accum<<<cdiv((long)Nn * 128, 256), 256, 0, stream>>>(hout, batch, pool, cnt, Nn, 128);
    k_pool_clf<<<cdiv(G * 3, 128), 128, 0, stream>>>(pool, cnt, clf_w, clf_b, out3, G, 128);
  };

  // 1-2) two GAT branches + their classifier heads
  run_branch(face_x, face_ei, EF, NF, face_b, fp, w_fin, w_fwl, w_fwr,
             clf_f_w, clf_f_b, h, hbf, out);
  run_branch(ctx_x, ctx_ei, EC, NC, ctx_b, cp, w_cin, w_cwl, w_cwr,
             clf_c_w, clf_c_b, h + (size_t)NF * 128, hbf + (size_t)NF * 128, out + G * 3);

  // 3) rn = concat(hf,hc) @ proj_w^T + proj_b   (f32 + bf16 shadow)
  gemm(stream, hbf, 128, w_proj, 128, proj_b, nullptr, nullptr, 0, nullptr,
       rn, rnbf, 256, (int)NA, 256, 128, false);

  // 4) combined batch + priors
  hipMemcpyAsync(rb, face_b, (size_t)NF * 4, hipMemcpyDeviceToDevice, stream);
  hipMemcpyAsync(rb + NF, ctx_b, (size_t)NC * 4, hipMemcpyDeviceToDevice, stream);
  hipMemcpyAsync(pr, face_pr, (size_t)NF * 4, hipMemcpyDeviceToDevice, stream);
  hipMemcpyAsync(pr + NF, ctx_pr, (size_t)NC * 4, hipMemcpyDeviceToDevice, stream);

  // 5) router hidden: relu(rn @ W1[:, :256]^T + priors * W1[:,256] + b1)
  gemm(stream, rnbf, 256, w_r1, 256, router_b1, pr, router_w1 + 256, 257, nullptr,
       nullptr, h1bf, 256, (int)NA, 256, 256, true);
  // 6) logits
  gemm(stream, h1bf, 256, w_r2, 256, router_b2, nullptr, nullptr, 0, nullptr,
       logits, nullptr, 4, (int)NA, 4, 256, false);
  // 7) hard top-3 routing mask
  k_topk<<<cdiv(NA, 256), 256, 0, stream>>>(logits, wm, NA);
  // 8) node messages
  gemm(stream, rnbf, 256, w_msg, 256, msg_b, nullptr, nullptr, 0, nullptr,
       r2v, nullptr, 256, (int)NA, 256, 256, false);
  // 9-10) scatter to virtual nodes, add init
  k_fill<<<cdiv((long)G * 1024, 256), 256, 0, stream>>>(msgs, 0.0f, (long)G * 1024);
  k_vn_scatter<<<cdiv(NA * 256, 256), 256, 0, stream>>>(r2v, wm, rb, msgs, NA);
  k_vn_build<<<cdiv((long)G * 1024, 256), 256, 0, stream>>>(vn, vnbf, vn_init, msgs, G);
  // 11-13) MHA over 4 VNs + residual
  gemm(stream, vnbf, 256, w_min, 256, mha_in_b, nullptr, nullptr, 0, nullptr,
       qkv, nullptr, 768, G * 4, 768, 256, false);
  k_mha<<<cdiv(G * 16, 128), 128, 0, stream>>>(qkv, attno, attnbf, G);
  gemm(stream, attnbf, 256, w_mout, 256, mha_out_b, nullptr, nullptr, 0, vn,
       vn2, nullptr, 256, G * 4, 256, 256, false);
  // 14) gather routed VN state back to nodes
  k_realmsg<<<cdiv(NA * 256, 256), 256, 0, stream>>>(vn2, wm, rb, rm, rmbf, NA);
  // 15-17) GRU cell
  gemm(stream, rmbf, 256, w_gih, 256, gru_b_ih, nullptr, nullptr, 0, nullptr,
       gi, nullptr, 768, (int)NA, 768, 256, false);
  gemm(stream, rnbf, 256, w_ghh, 256, gru_b_hh, nullptr, nullptr, 0, nullptr,
       gh, nullptr, 768, (int)NA, 768, 256, false);
  k_gru<<<cdiv(NA * 256, 256), 256, 0, stream>>>(gi, gh, rn, rm, NA);  // rm := updated
  // 18) whole-graph head
  k_fill<<<cdiv((long)G * 256, 256), 256, 0, stream>>>(pool, 0.0f, (long)G * 256);
  k_fill<<<cdiv((long)G, 256), 256, 0, stream>>>(cnt, 0.0f, (long)G);
  k_pool_accum<<<cdiv(NA * 256, 256), 256, 0, stream>>>(rm, rb, pool, cnt, NA, 256);
  k_pool_clf<<<cdiv(G * 3, 128), 128, 0, stream>>>(pool, cnt, clf_w_w, clf_w_b, out + 2 * G * 3, G, 256);
  (void)n_in; (void)out_size; (void)ws_size;
}
